// KNearestNeighborEntropyLoss_22024592294535
// MI455X (gfx1250) — matrix operand order repaired and checked by reading, not compiled, not run on hardware
//
#include <hip/hip_runtime.h>
#include <hip/hip_bf16.h>
#include <math.h>

typedef _Float16 half_t;
typedef __attribute__((ext_vector_type(16))) _Float16 v16h_t;
typedef __attribute__((ext_vector_type(8)))  _Float16 v8h_t;
typedef __attribute__((ext_vector_type(8)))  float    v8f_t;
typedef int v4i_t __attribute__((vector_size(16)));   // matches builtin param type

#define B_N    8192
#define D_K    768
#define MT     128                         // rows per workgroup
#define NT     64                          // cols per chunk
#define PK     128                         // K panel resident in LDS
#define KP_PER_CHUNK (D_K / PK)            // 6 panels per chunk
#define NSLICE 8                           // column slices (grid.y)
#define COLS_PER_SLICE   (B_N / NSLICE)    // 1024
#define CHUNKS_PER_SLICE (COLS_PER_SLICE / NT) // 16
#define NPANELS (CHUNKS_PER_SLICE * KP_PER_CHUNK) // 96
#define INF_F  3.0e38f

// CDNA5 async memory->LDS path (ASYNCcnt-tracked, no VGPR round trip).
#if defined(__HIP_DEVICE_COMPILE__) && defined(__gfx1250__) && \
    __has_builtin(__builtin_amdgcn_global_load_async_to_lds_b128)
#define USE_ASYNC_LDS 1
#else
#define USE_ASYNC_LDS 0
#endif

#if USE_ASYNC_LDS
typedef __attribute__((address_space(1))) v4i_t* gv4i_p;
typedef __attribute__((address_space(3))) v4i_t* lv4i_p;
#if __has_builtin(__builtin_amdgcn_s_wait_asynccnt)
#define WAIT_ASYNC() __builtin_amdgcn_s_wait_asynccnt(0)
#else
#define WAIT_ASYNC() asm volatile("s_wait_asynccnt 0x0" ::: "memory")
#endif
#else
#define WAIT_ASYNC() do { } while (0)
#endif

// Keep k[0..4] sorted ascending; insert v if it beats the current max.
__device__ __forceinline__ void insert5(float k[5], float v) {
  if (v < k[4]) {
    k[4] = v;
    float t;
    if (k[4] < k[3]) { t = k[3]; k[3] = k[4]; k[4] = t; }
    if (k[3] < k[2]) { t = k[2]; k[2] = k[3]; k[3] = t; }
    if (k[2] < k[1]) { t = k[1]; k[1] = k[2]; k[2] = t; }
    if (k[1] < k[0]) { t = k[0]; k[0] = k[1]; k[1] = t; }
  }
}

__global__ void zero_out_kernel(float* __restrict__ out, int n) {
  int i = blockIdx.x * blockDim.x + threadIdx.x;
  if (i < n) out[i] = 0.0f;
}

// Phase 0: fp32 -> fp16 copy + row squared norms.
__global__ __launch_bounds__(256)
void knn_prep(const float* __restrict__ x, half_t* __restrict__ xh,
              float* __restrict__ sq) {
  __shared__ float red[256];
  const int r = blockIdx.x;
  const int t = threadIdx.x;
  const float* row  = x  + (size_t)r * D_K;
  half_t*      hrow = xh + (size_t)r * D_K;
  float s = 0.0f;
  for (int i = t; i < D_K; i += 256) {
    float v = row[i];
    s += v * v;
    hrow[i] = (half_t)v;
  }
  red[t] = s;
  __syncthreads();
  for (int off = 128; off > 0; off >>= 1) {
    if (t < off) red[t] += red[t + off];
    __syncthreads();
  }
  if (t == 0) sq[r] = red[0];
}

// Issue the fill of one 64-col x 128-K B panel (16 KB) into an LDS buffer.
// Async path: per-lane GLOBAL_LOAD_ASYNC_TO_LDS_B128 (4 per thread).
__device__ __forceinline__ void issue_panel(const half_t* __restrict__ xh,
                                            int sliceCol, int p,
                                            half_t* sBbuf, int tid) {
  const int chunk   = p / KP_PER_CHUNK;
  const int kp      = (p - chunk * KP_PER_CHUNK) * PK;
  const int colBase = sliceCol + chunk * NT;
#pragma unroll
  for (int i = 0; i < 4; ++i) {
    const int idx  = i * 256 + tid;
    const int bcol = idx >> 4;
    const int seg  = idx & 15;
    const half_t* g = xh + (size_t)(colBase + bcol) * D_K + kp + seg * 8;
    half_t*       l = sBbuf + bcol * PK + seg * 8;
#if USE_ASYNC_LDS
    __builtin_amdgcn_global_load_async_to_lds_b128(
        (gv4i_p)g, (lv4i_p)l, 0, 0);
#else
    *(v8h_t*)l = *(const v8h_t*)g;
#endif
  }
}

// Phase 1: fused WMMA Gram tile -> distances -> streaming per-row top-5.
// grid = (B_N/MT, NSLICE), block = 256 (8 wave32s).
__global__ __launch_bounds__(256)
void knn_gemm_topk(const half_t* __restrict__ xh, const float* __restrict__ sq,
                   float* __restrict__ cand) {
  __shared__ __align__(16) half_t sB[2][NT * PK];  // 2 x 16 KB double-buffered B panel
  __shared__ __align__(16) float  sDist[MT * NT];  // 32 KB distance tile

  const int tid  = threadIdx.x;
  const int wave = tid >> 5;      // 0..7, one 16-row strip each
  const int lane = tid & 31;
  const int l16  = lane & 15;
  const int lhi  = lane >> 4;     // 0/1

  const int rowBlock = blockIdx.x * MT;
  const int slice    = blockIdx.y;
  const int sliceCol = slice * COLS_PER_SLICE;

  const int waveRow = rowBlock + wave * 16;
  // A fragment source row for this lane (ISA 16-bit A 16x32 layout: M = lane&15)
  const half_t* aRow = xh + (size_t)(waveRow + l16) * D_K;

  // sq for the 8 output rows this lane produces: M = v + 8*lhi (C/D layout)
  float sqr[8];
#pragma unroll
  for (int v = 0; v < 8; ++v) sqr[v] = sq[waveRow + v + 8 * lhi];

  // streaming top-5 state: thread owns (row tid>>1, column half tid&1)
  float tk[5] = { INF_F, INF_F, INF_F, INF_F, INF_F };
  const int myRow  = tid >> 1;
  const int myPart = tid & 1;

  // pipeline prologue: start DMA of panel 0
  issue_panel(xh, sliceCol, 0, sB[0], tid);

  for (int chunk = 0; chunk < CHUNKS_PER_SLICE; ++chunk) {
    const int colBase = sliceCol + chunk * NT;

    v8f_t acc[4];
#pragma unroll
    for (int t = 0; t < 4; ++t) { v8f_t z = {}; acc[t] = z; }

    float sqc[4];
#pragma unroll
    for (int t = 0; t < 4; ++t) sqc[t] = sq[colBase + t * 16 + l16];

    for (int kpi = 0; kpi < KP_PER_CHUNK; ++kpi) {
      const int p = chunk * KP_PER_CHUNK + kpi;
      WAIT_ASYNC();          // own async ops for panel p complete (in-order)
      __syncthreads();       // all waves' panel-p LDS writes visible; buf (p+1)&1 free
      if (p + 1 < NPANELS)   // overlap next panel's DMA with this panel's WMMAs
        issue_panel(xh, sliceCol, p + 1, sB[(p + 1) & 1], tid);

      const half_t* sBcur = sB[p & 1];
      const int kp = kpi * PK;
#pragma unroll
      for (int ks = 0; ks < PK; ks += 32) {
        const int k0 = kp + ks;
        // A frag: lanes 0-15 hold K {k0..k0+7, k0+16..k0+23}; lanes 16-31 shift by 8
        v8h_t a_lo = *(const v8h_t*)(aRow + k0 + lhi * 8);
        v8h_t a_hi = *(const v8h_t*)(aRow + k0 + lhi * 8 + 16);
        v16h_t afrag = __builtin_shufflevector(a_lo, a_hi,
            0,1,2,3,4,5,6,7,8,9,10,11,12,13,14,15);
        // load all four B frags first so ds_loads clause and WMMAs go back-to-back
        v16h_t bfrag[4];
#pragma unroll
        for (int t = 0; t < 4; ++t) {
          const half_t* bp = sBcur + (t * 16 + l16) * PK + ks + lhi * 8;
          v8h_t b_lo = *(const v8h_t*)bp;
          v8h_t b_hi = *(const v8h_t*)(bp + 16);
          bfrag[t] = __builtin_shufflevector(b_lo, b_hi,
              0,1,2,3,4,5,6,7,8,9,10,11,12,13,14,15);
        }
#pragma unroll
        for (int t = 0; t < 4; ++t) {
          acc[t] = __builtin_amdgcn_wmma_f32_16x16x32_f16(
              false, afrag, false, bfrag[t], (short)0, acc[t], false, false);
        }
      }
    }

    // distances -> LDS tile (C/D layout: VGPR v -> M = v + 8*lhi, N = lane&15)
#pragma unroll
    for (int t = 0; t < 4; ++t) {
#pragma unroll
      for (int v = 0; v < 8; ++v) {
        float g  = acc[t][v];
        int   rl = wave * 16 + v + 8 * lhi;
        int   cl = t * 16 + l16;
        float d2 = sqr[v] + sqc[t] - 2.0f * g;
        float d  = d2 > 0.0f ? sqrtf(d2) : 0.0f;
        if (rowBlock + rl == colBase + cl) d = INF_F;  // mask self
        sDist[rl * NT + cl] = d;
      }
    }
    __syncthreads();

    // streaming top-5 update: 2 threads per row, 32 columns each.
    // Rotate scan start by (row & 31) so the 32 lanes of a wave hit 32
    // distinct LDS banks (unrotated access is 16-way conflicted).
    const float* rowd = sDist + myRow * NT + myPart * 32;
    const int rot = myRow & 31;
#pragma unroll 8
    for (int c = 0; c < 32; ++c) insert5(tk, rowd[(c + rot) & 31]);
    __syncthreads();
  }

  // merge two per-row partial top-5 lists (reuse sDist)
  if (myPart == 1) {
#pragma unroll
    for (int i = 0; i < 5; ++i) sDist[myRow * 5 + i] = tk[i];
  }
  __syncthreads();
  if (myPart == 0) {
#pragma unroll
    for (int i = 0; i < 5; ++i) insert5(tk, sDist[myRow * 5 + i]);
    float* c = cand + ((size_t)(rowBlock + myRow) * NSLICE + slice) * 5;
#pragma unroll
    for (int i = 0; i < 5; ++i) c[i] = tk[i];
  }
}

// Phase 2: per-row merge of per-slice candidates, loss reduction.
__global__ __launch_bounds__(256)
void knn_loss(const float* __restrict__ cand, float* __restrict__ out) {
  int r = blockIdx.x * blockDim.x + threadIdx.x;
  if (r >= B_N) return;
  const float* c = cand + (size_t)r * (NSLICE * 5);
  float tk[5] = { INF_F, INF_F, INF_F, INF_F, INF_F };
#pragma unroll
  for (int i = 0; i < NSLICE * 5; ++i) insert5(tk, c[i]);
  float mean    = (tk[0] + tk[1] + tk[2] + tk[3] + tk[4]) * 0.2f;
  float contrib = -logf(mean + 1e-8f) * (1.0f / (float)B_N);
  atomicAdd(out, contrib);
}

extern "C" void kernel_launch(void* const* d_in, const int* in_sizes, int n_in,
                              void* d_out, int out_size, void* d_ws, size_t ws_size,
                              hipStream_t stream) {
  (void)in_sizes; (void)n_in; (void)ws_size;
  const float* x   = (const float*)d_in[0];
  float*       out = (float*)d_out;

  // workspace layout: fp16 copy (12.6 MB) | row norms (32 KB) | candidates (1.3 MB)
  half_t* xh  = (half_t*)d_ws;
  size_t  off = (size_t)B_N * D_K * sizeof(half_t);
  float*  sq  = (float*)((char*)d_ws + off);
  off += (size_t)B_N * sizeof(float);
  float*  cand = (float*)((char*)d_ws + off);

  zero_out_kernel<<<1, 64, 0, stream>>>(out, out_size);
  knn_prep<<<B_N, 256, 0, stream>>>(x, xh, sq);
  dim3 grid(B_N / MT, NSLICE);
  knn_gemm_topk<<<grid, 256, 0, stream>>>(xh, sq, cand);
  knn_loss<<<B_N / 256, 256, 0, stream>>>(cand, out);
}
